// TTTBlock_75754633167132
// MI455X (gfx1250) — matrix operand order repaired and checked by reading, not compiled, not run on hardware
//
#include <hip/hip_runtime.h>
#include <hip/hip_bf16.h>
#include <cstdint>

// ---------------------------------------------------------------------------
// Types for CDNA5 WMMA (wave32): v_wmma_f32_16x16x32_bf16
// ---------------------------------------------------------------------------
typedef __attribute__((ext_vector_type(16))) __bf16 v16bf;
typedef __attribute__((ext_vector_type(8)))  float  v8f;

union FragU {
    uint4 u4[2];
    v16bf v;
};
static_assert(sizeof(FragU) == 32, "frag size");

// ---------------------------------------------------------------------------
// fp32 -> bf16 (RNE) split helpers for bf16x3 fp32-emulation
// ---------------------------------------------------------------------------
__device__ __forceinline__ uint32_t f32_to_bf16(float f) {
    uint32_t u = __float_as_uint(f);
    uint32_t r = u + 0x7FFFu + ((u >> 16) & 1u);
    return r >> 16;
}
__device__ __forceinline__ float bf16_to_f32(uint32_t h) {
    return __uint_as_float(h << 16);
}

// Convert 8 consecutive fp32 into packed-bf16 hi (uint4) and lo (uint4).
__device__ __forceinline__ void cvt8(const float4& f0, const float4& f1,
                                     uint4& hi, uint4& lo) {
    float f[8] = {f0.x, f0.y, f0.z, f0.w, f1.x, f1.y, f1.z, f1.w};
    uint32_t h[8], l[8];
#pragma unroll
    for (int i = 0; i < 8; ++i) {
        h[i] = f32_to_bf16(f[i]);
        float r = f[i] - bf16_to_f32(h[i]);
        l[i] = f32_to_bf16(r);
    }
    hi.x = h[0] | (h[1] << 16); hi.y = h[2] | (h[3] << 16);
    hi.z = h[4] | (h[5] << 16); hi.w = h[6] | (h[7] << 16);
    lo.x = l[0] | (l[1] << 16); lo.y = l[2] | (l[3] << 16);
    lo.z = l[4] | (l[5] << 16); lo.w = l[6] | (l[7] << 16);
}

__device__ __forceinline__ v16bf load_frag(const uint16_t* b0, const uint16_t* b1) {
    FragU f;
    f.u4[0] = *(const uint4*)b0;
    f.u4[1] = *(const uint4*)b1;
    return f.v;
}

// ---------------------------------------------------------------------------
// LayerNorm: one 1024-wide row per 256-thread block
// ---------------------------------------------------------------------------
__global__ __launch_bounds__(256) void ln_kernel(const float* __restrict__ x,
                                                 const float* __restrict__ g,
                                                 const float* __restrict__ b,
                                                 float* __restrict__ out) {
    const int C = 1024;
    size_t row = blockIdx.x;
    int tid = threadIdx.x;
    const float* xr = x + row * C;
    float4 xv = *(const float4*)(xr + tid * 4);
    float s  = xv.x + xv.y + xv.z + xv.w;
    float s2 = xv.x * xv.x + xv.y * xv.y + xv.z * xv.z + xv.w * xv.w;
#pragma unroll
    for (int m = 1; m < 32; m <<= 1) {
        s  += __shfl_xor(s,  m, 32);
        s2 += __shfl_xor(s2, m, 32);
    }
    __shared__ float ps[8], ps2[8];
    if ((tid & 31) == 0) { ps[tid >> 5] = s; ps2[tid >> 5] = s2; }
    __syncthreads();
    s = 0.f; s2 = 0.f;
#pragma unroll
    for (int i = 0; i < 8; ++i) { s += ps[i]; s2 += ps2[i]; }
    float mean = s * (1.0f / C);
    float var  = s2 * (1.0f / C) - mean * mean;
    float rstd = rsqrtf(var + 1e-5f);
    float4 gv = *(const float4*)(g + tid * 4);
    float4 bv = *(const float4*)(b + tid * 4);
    float4 o;
    o.x = (xv.x - mean) * rstd * gv.x + bv.x;
    o.y = (xv.y - mean) * rstd * gv.y + bv.y;
    o.z = (xv.z - mean) * rstd * gv.z + bv.z;
    o.w = (xv.w - mean) * rstd * gv.w + bv.w;
    *(float4*)(out + row * C + tid * 4) = o;
}

// ---------------------------------------------------------------------------
// GEMM: out[M,N] = epi( A[M,K] @ W[N,K]^T + bias ) + resid
// bf16x3 emulated-fp32 via v_wmma_f32_16x16x32_bf16, fp32 accumulate.
// Block = 256 threads (8 waves), tile 64x128, BK=32, wave tile 32x32.
// ---------------------------------------------------------------------------
#define BM 64
#define BN 128
#define BK 32
#define LDT 40   // padded LDS row stride in bf16 elements (80B: conflict-free)

__global__ __launch_bounds__(256) void gemm_bf16x3(
        const float* __restrict__ A, const float* __restrict__ Wt,
        const float* __restrict__ bias, const float* __restrict__ resid,
        float* __restrict__ out, int M, int N, int K, int fuseGelu) {
    __shared__ alignas(16) uint16_t sAh[BM * LDT];
    __shared__ alignas(16) uint16_t sAl[BM * LDT];
    __shared__ alignas(16) uint16_t sBh[BN * LDT];
    __shared__ alignas(16) uint16_t sBl[BN * LDT];

    int tid  = threadIdx.x;
    int lane = tid & 31;
    int wv   = tid >> 5;
    int wm   = wv >> 2;          // 0..1
    int wn   = wv & 3;           // 0..3
    int m0   = blockIdx.y * BM;
    int n0   = blockIdx.x * BN;

    // global staging assignments
    int ar = tid >> 2, ac = (tid & 3) * 8;     // A: 64 rows x 32, 8 floats/thread
    int br = tid >> 1, bc = (tid & 1) * 16;    // B: 128 rows x 32, 16 floats/thread
    const float* gA = A  + (size_t)(m0 + ar) * K + ac;
    const float* gB = Wt + (size_t)(n0 + br) * K + bc;

    // fragment addressing (ISA 7.12.2 layouts)
    int hh    = lane >> 4;       // lane half
    int mrow  = lane & 15;
    int aoff0 = hh * 8;          // A VGPR0-3 K-chunk
    int aoff1 = 16 + hh * 8;     // A VGPR4-7 K-chunk
    int boff  = hh * 16;         // unused; B frags load via aoff0/aoff1 like A

    v8f acc[2][2] = {};

    for (int k0 = 0; k0 < K; k0 += BK) {
        float4 a0 = *(const float4*)(gA + k0);
        float4 a1 = *(const float4*)(gA + k0 + 4);
        float4 b0 = *(const float4*)(gB + k0);
        float4 b1 = *(const float4*)(gB + k0 + 4);
        float4 b2 = *(const float4*)(gB + k0 + 8);
        float4 b3 = *(const float4*)(gB + k0 + 12);
        // speculative prefetch of next k-slice (global_prefetch_b8)
        __builtin_prefetch(gA + k0 + BK, 0, 0);
        __builtin_prefetch(gB + k0 + BK, 0, 0);

        __syncthreads();
        uint4 hi, lo;
        cvt8(a0, a1, hi, lo);
        *(uint4*)&sAh[ar * LDT + ac] = hi;
        *(uint4*)&sAl[ar * LDT + ac] = lo;
        cvt8(b0, b1, hi, lo);
        *(uint4*)&sBh[br * LDT + bc] = hi;
        *(uint4*)&sBl[br * LDT + bc] = lo;
        cvt8(b2, b3, hi, lo);
        *(uint4*)&sBh[br * LDT + bc + 8] = hi;
        *(uint4*)&sBl[br * LDT + bc + 8] = lo;
        __syncthreads();

        v16bf ah[2], al[2], bh[2], bl[2];
#pragma unroll
        for (int mt = 0; mt < 2; ++mt) {
            int m = (wm * 32 + mt * 16 + mrow) * LDT;
            ah[mt] = load_frag(&sAh[m + aoff0], &sAh[m + aoff1]);
            al[mt] = load_frag(&sAl[m + aoff0], &sAl[m + aoff1]);
        }
#pragma unroll
        for (int nt = 0; nt < 2; ++nt) {
            int n = (wn * 32 + nt * 16 + mrow) * LDT;
            bh[nt] = load_frag(&sBh[n + aoff0], &sBh[n + aoff1]);
            bl[nt] = load_frag(&sBl[n + aoff0], &sBl[n + aoff1]);
        }
#pragma unroll
        for (int mt = 0; mt < 2; ++mt)
#pragma unroll
            for (int nt = 0; nt < 2; ++nt) {
                acc[mt][nt] = __builtin_amdgcn_wmma_f32_16x16x32_bf16(
                    false, al[mt], false, bh[nt], (short)0, acc[mt][nt], false, false);
                acc[mt][nt] = __builtin_amdgcn_wmma_f32_16x16x32_bf16(
                    false, ah[mt], false, bl[nt], (short)0, acc[mt][nt], false, false);
                acc[mt][nt] = __builtin_amdgcn_wmma_f32_16x16x32_bf16(
                    false, ah[mt], false, bh[nt], (short)0, acc[mt][nt], false, false);
            }
    }

    // Epilogue. C/D layout: VGPR v -> row = h*8 + v, col = lane%16.
#pragma unroll
    for (int mt = 0; mt < 2; ++mt) {
#pragma unroll
        for (int nt = 0; nt < 2; ++nt) {
            int col = n0 + wn * 32 + nt * 16 + mrow;
            float bz = bias ? bias[col] : 0.0f;
#pragma unroll
            for (int v = 0; v < 8; ++v) {
                int row = m0 + wm * 32 + mt * 16 + hh * 8 + v;
                float val = acc[mt][nt][v] + bz;
                if (fuseGelu)
                    val = 0.5f * val * (1.0f + erff(val * 0.70710678118654752f));
                if (resid) val += resid[(size_t)row * N + col];
                out[(size_t)row * N + col] = val;
            }
        }
    }
}

// ---------------------------------------------------------------------------
// TTT scan. W rows are independent across the whole scan:
//   err_i = w_i.k - v_i ; w_i -= lr*err_i*k ; out_i = w_i.q (pre-update)
// One wave per W row (2 fp32/lane), 8 rows per block, 512 blocks total.
// ---------------------------------------------------------------------------
__global__ __launch_bounds__(256) void ttt_scan(const float* __restrict__ q,
                                                const float* __restrict__ k,
                                                const float* __restrict__ v,
                                                const float* __restrict__ lr_scale,
                                                float* __restrict__ out) {
    const int H = 16, D = 64, T = 2048;
    int bh = blockIdx.x >> 3;            // 0..63 = b*16+h
    int rg = blockIdx.x & 7;             // row group (8 rows)
    int h  = bh & 15;
    int tid = threadIdx.x;
    int wave = tid >> 5, lane = tid & 31;
    int i  = rg * 8 + wave;              // W row 0..63
    int j0 = lane * 2;

    float eff = 0.01f * lr_scale[h];
    float w0 = 0.0f, w1 = 0.0f;

    __shared__ float sq[64], sk[64], sv8[8];
    size_t base = (size_t)bh * D;        // ((b*T + t)*H + h)*D with C=H*D
    const size_t step = (size_t)H * D;

    for (int t = 0; t < T; ++t) {
        size_t off = base + (size_t)t * step * 4 / 4; // base + t*H*D within (b) chunk
        // layout index(b,t,h,d) = b*T*H*D + t*H*D + h*D + d ; base holds b*?:
        // recompute exactly:
        off = ((size_t)(bh >> 4) * T + t) * step + (size_t)h * D;
        if (tid < 64)        sq[tid]        = q[off + tid];
        else if (tid < 128)  sk[tid - 64]   = k[off + tid - 64];
        else if (tid < 136)  sv8[tid - 128] = v[off + rg * 8 + (tid - 128)];
        __syncthreads();

        float qa = sq[j0], qb = sq[j0 + 1];
        float ka = sk[j0], kb = sk[j0 + 1];
        float dq = w0 * qa + w1 * qb;
        float dk = w0 * ka + w1 * kb;
#pragma unroll
        for (int m = 1; m < 32; m <<= 1) {
            dq += __shfl_xor(dq, m, 32);
            dk += __shfl_xor(dk, m, 32);
        }
        float err = dk - sv8[wave];
        w0 -= eff * err * ka;
        w1 -= eff * err * kb;
        if (lane == 0) out[off + i] = dq;    // read-before-update semantics
        __syncthreads();
    }
}

// ---------------------------------------------------------------------------
// Launch
// ---------------------------------------------------------------------------
extern "C" void kernel_launch(void* const* d_in, const int* in_sizes, int n_in,
                              void* d_out, int out_size, void* d_ws, size_t ws_size,
                              hipStream_t stream) {
    const float* x    = (const float*)d_in[0];
    const float* ln1w = (const float*)d_in[1];
    const float* ln1b = (const float*)d_in[2];
    const float* wq   = (const float*)d_in[3];
    const float* wk   = (const float*)d_in[4];
    const float* wvp  = (const float*)d_in[5];
    const float* wo   = (const float*)d_in[6];
    const float* lrs  = (const float*)d_in[7];
    const float* ln2w = (const float*)d_in[8];
    const float* ln2b = (const float*)d_in[9];
    const float* w1   = (const float*)d_in[10];
    const float* b1   = (const float*)d_in[11];
    const float* w2   = (const float*)d_in[12];
    const float* b2   = (const float*)d_in[13];
    float* out = (float*)d_out;
    float* ws  = (float*)d_ws;

    const int Bb = 4, T = 2048, C = 1024;
    const int M = Bb * T;                 // 8192
    const size_t MC = (size_t)M * C;      // 8M floats

    float* xn  = ws;                      // [M,C]   (reused as ln2 output)
    float* qb  = ws + MC;                 // [M,C]
    float* kb  = ws + 2 * MC;             // [M,C]
    float* vb  = ws + 3 * MC;             // [M,C]
    float* tt  = ws + 4 * MC;             // [M,C]
    float* hb  = qb;                      // [M,4C] aliases q/k/v/tt (dead by then)

    dim3 blk(256);
    dim3 gLN(M);
    dim3 gN1(C / BN, M / BM);             // (8, 128)
    dim3 gN4(4 * C / BN, M / BM);         // (32, 128)

    // x_n = LN1(x)
    ln_kernel<<<gLN, blk, 0, stream>>>(x, ln1w, ln1b, xn);
    // q,k,v = x_n @ w{q,k,v}^T
    gemm_bf16x3<<<gN1, blk, 0, stream>>>(xn, wq,  nullptr, nullptr, qb, M, C, C, 0);
    gemm_bf16x3<<<gN1, blk, 0, stream>>>(xn, wk,  nullptr, nullptr, kb, M, C, C, 0);
    gemm_bf16x3<<<gN1, blk, 0, stream>>>(xn, wvp, nullptr, nullptr, vb, M, C, C, 0);
    // sequential fast-weight scan (row-parallel)
    ttt_scan<<<dim3(512), blk, 0, stream>>>(qb, kb, vb, lrs, tt);
    // x2 = x + ttt @ wo^T   (into d_out)
    gemm_bf16x3<<<gN1, blk, 0, stream>>>(tt, wo, nullptr, x, out, M, C, C, 0);
    // h_n = LN2(x2)
    ln_kernel<<<gLN, blk, 0, stream>>>(out, ln2w, ln2b, xn);
    // h = gelu(h_n @ w1^T + b1)
    gemm_bf16x3<<<gN4, blk, 0, stream>>>(xn, w1, b1, nullptr, hb, M, 4 * C, C, 1);
    // out = x2 + h @ w2^T + b2
    gemm_bf16x3<<<gN1, blk, 0, stream>>>(hb, w2, b2, out, out, M, C, 4 * C, 0);
}
